// SimpleGRU_37830071943830
// MI455X (gfx1250) — compile-verified, matrix operand order checked by reading, not run on hardware
//
#include <hip/hip_runtime.h>
#include <hip/hip_bf16.h>

// ---------------------------------------------------------------------------
// GRU (T=512,B=64,I=512,H=1024) for gfx1250 (MI455X, wave32, WMMA).
//
// Phase 0: convert x -> bf16; repack [w_ih | w_hh] into *fragment-major* bf16
//          layout: for each (jtile, gate, kstep) the exact 32-lane x 32B
//          WMMA B-fragment is stored contiguously (1KB), so the recurrence's
//          B loads are fully coalesced b128s with immediate offsets.
// Phase 1: persistent cooperative kernel, 32 blocks x 256 threads.
//          Each wave owns a 16x16 output tile (16 batch rows x 16 hidden
//          units) and carries 4 f32 WMMA accumulators: r, z, n_x, n_h
//          (n split at K=512 because r multiplies only the h-part).
//          Grid barrier (atomic + fences) once per timestep.
// ---------------------------------------------------------------------------

typedef __bf16 bf16_t;
typedef __attribute__((ext_vector_type(16))) __bf16 v16bf;
typedef __attribute__((ext_vector_type(8)))  __bf16 v8bf;
typedef __attribute__((ext_vector_type(8)))  float  v8f;

#define T_LEN   512
#define B_DIM   64
#define I_DIM   512
#define H_DIM   1024
#define K_CAT   (I_DIM + H_DIM)      // 1536
#define KSTEPS  (K_CAT / 32)         // 48 wmma k-steps
#define KSTEPS_X (I_DIM / 32)        // 16 k-steps covering the x-part
#define NBLOCKS 32
#define NTHREADS 256
#define NJT     (H_DIM / 16)         // 64 j-tiles
#define LDS_STRIDE (K_CAT + 8)       // 1544 bf16 -> 3088B row stride, bank-conflict free
#define FRAG_ELEMS 512               // 32 lanes x 16 bf16 per B-fragment (1KB)

// Workspace layout (bytes)
#define CTR_OFF   0u
#define HBUF_OFF  1024u                                   // 2 x (B*H) bf16 ping-pong
#define HBUF_ONE  ((size_t)B_DIM * H_DIM * 2)             // 131072
#define WPK_OFF   (HBUF_OFF + 2 * (unsigned)HBUF_ONE)     // 263168 (1KB aligned)
#define WPK_ELEMS ((size_t)NJT * 3 * KSTEPS * FRAG_ELEMS) // 4718592 (== 3H * K_CAT)
#define XBF_OFF   (WPK_OFF + (unsigned)(WPK_ELEMS * 2))   // 9700352 (1KB aligned)
#define X_ELEMS   ((size_t)T_LEN * B_DIM * I_DIM)         // 16777216

union V16U { v16bf v; v8bf h[2]; };

static __device__ __forceinline__
v8f wmma_bf16(v8bf alo, v8bf ahi, v8bf blo, v8bf bhi, v8f acc) {
    V16U a, b;
    a.h[0] = alo; a.h[1] = ahi;
    b.h[0] = blo; b.h[1] = bhi;
    return __builtin_amdgcn_wmma_f32_16x16x32_bf16(
        /*neg_a=*/false, a.v, /*neg_b=*/false, b.v,
        /*c_mod=*/(short)0, acc, /*reuse_a=*/false, /*reuse_b=*/false);
}

static __device__ __forceinline__ float fast_sigmoid(float x) {
    return 1.0f / (1.0f + __expf(-x));
}
static __device__ __forceinline__ float fast_tanh(float x) {
    float ax = fabsf(x);
    float e  = __expf(-2.0f * ax);
    float t  = (1.0f - e) / (1.0f + e);
    return copysignf(t, x);
}

// ---- Phase 0 kernels -------------------------------------------------------

__global__ void cvt_x_kernel(const float* __restrict__ x, bf16_t* __restrict__ xb) {
    size_t i = (size_t)blockIdx.x * blockDim.x + threadIdx.x;
    size_t stride = (size_t)gridDim.x * blockDim.x;
    for (; i < X_ELEMS; i += stride)
        xb[i] = (bf16_t)x[i];
}

// Fragment-major weight repack.
// wpk[(((jt*3 + g)*KSTEPS + k0)*32 + lane)*16 + e]:
//   half = e/8 (0=lo,1=hi), ei = e%8
//   j    = jt*16 + (lane & 15)
//   k    = k0*32 + half*16 + (lane>>4)*8 + ei        (K over [x ; h])
//   src  = k < I_DIM ? w_ih[g*H + j][k] : w_hh[g*H + j][k - I_DIM]
__global__ void build_wpk_kernel(const float* __restrict__ wih,
                                 const float* __restrict__ whh,
                                 bf16_t* __restrict__ wpk) {
    size_t i = (size_t)blockIdx.x * blockDim.x + threadIdx.x;
    size_t stride = (size_t)gridDim.x * blockDim.x;
    for (; i < WPK_ELEMS; i += stride) {
        int e    = (int)(i & 15);
        int lane = (int)((i >> 4) & 31);
        int k0   = (int)((i >> 9) % KSTEPS);
        size_t q = i / ((size_t)KSTEPS << 9);
        int g    = (int)(q % 3);
        int jt   = (int)(q / 3);

        int j = jt * 16 + (lane & 15);
        int k = k0 * 32 + (e >> 3) * 16 + ((lane >> 4) << 3) + (e & 7);
        size_t row = (size_t)g * H_DIM + j;
        float v = (k < I_DIM) ? wih[row * I_DIM + k]
                              : whh[row * H_DIM + (k - I_DIM)];
        wpk[i] = (bf16_t)v;
    }
}

// ---- Phase 1: persistent recurrence ---------------------------------------

__global__ __launch_bounds__(NTHREADS)
void gru_scan_kernel(const bf16_t* __restrict__ xb,
                     const bf16_t* __restrict__ wpk,
                     const float*  __restrict__ bih,
                     const float*  __restrict__ bhh,
                     float*        __restrict__ out,
                     bf16_t*       __restrict__ hbuf,
                     unsigned*     __restrict__ ctr) {
    __shared__ bf16_t lds_a[16 * LDS_STRIDE];   // [x_t ; h] rows for this block

    const int tid   = threadIdx.x;
    const int lane  = tid & 31;
    const int wave  = tid >> 5;                 // 0..7
    const int mtile = blockIdx.x >> 3;          // 0..3  (16 batch rows)
    const int jgrp  = blockIdx.x & 7;           // 0..7  (128 hidden cols)
    const int m0    = mtile * 16;
    const int jtile = jgrp * 8 + wave;          // 0..63
    const int j0    = jtile * 16;

    const int am   = lane & 15;                 // A row within tile
    const int koff = (lane >> 4) * 8;           // K sub-offset per ISA layout
    const int j    = j0 + (lane & 15);          // this lane's hidden unit

    // Fragment-major B bases: advance by FRAG_ELEMS (512 bf16 = 1KB) per k-step.
    const bf16_t* pr = wpk + ((size_t)(jtile * 3 + 0) * KSTEPS) * FRAG_ELEMS + lane * 16;
    const bf16_t* pz = wpk + ((size_t)(jtile * 3 + 1) * KSTEPS) * FRAG_ELEMS + lane * 16;
    const bf16_t* pn = wpk + ((size_t)(jtile * 3 + 2) * KSTEPS) * FRAG_ELEMS + lane * 16;

    const float brc = bih[j] + bhh[j];
    const float bzc = bih[H_DIM + j] + bhh[H_DIM + j];
    const float bnx = bih[2 * H_DIM + j];
    const float bnh = bhh[2 * H_DIM + j];

    for (int t = 0; t < T_LEN; ++t) {
        const bf16_t* hprev = hbuf + (size_t)(t & 1) * ((size_t)B_DIM * H_DIM);
        bf16_t*       hnext = hbuf + (size_t)((t + 1) & 1) * ((size_t)B_DIM * H_DIM);

        // Stage A = [x_t ; h_prev] (16 rows x 1536 bf16) into LDS, 16B chunks.
        for (int c = tid; c < 16 * (K_CAT / 8); c += NTHREADS) {
            int row  = c / (K_CAT / 8);
            int col8 = c % (K_CAT / 8);
            v8bf d;
            if (col8 < I_DIM / 8)
                d = *(const v8bf*)(xb + ((size_t)t * B_DIM + m0 + row) * I_DIM + col8 * 8);
            else
                d = *(const v8bf*)(hprev + (size_t)(m0 + row) * H_DIM + (col8 - I_DIM / 8) * 8);
            *(v8bf*)(&lds_a[row * LDS_STRIDE + col8 * 8]) = d;
        }
        __syncthreads();

        v8f accr = {}, accz = {}, accnx = {}, accnh = {};
        const bf16_t* arow = &lds_a[am * LDS_STRIDE];

        // x-part of K (k0 = 0..15): gate-n accumulates into accnx.
        #pragma unroll 4
        for (int k0 = 0; k0 < KSTEPS_X; ++k0) {
            const int kk = k0 * 32;
            const size_t fo = (size_t)k0 * FRAG_ELEMS;
            v8bf alo = *(const v8bf*)(arow + kk + koff);
            v8bf ahi = *(const v8bf*)(arow + kk + 16 + koff);
            v8bf rlo = *(const v8bf*)(pr + fo);
            v8bf rhi = *(const v8bf*)(pr + fo + 8);
            v8bf zlo = *(const v8bf*)(pz + fo);
            v8bf zhi = *(const v8bf*)(pz + fo + 8);
            v8bf nlo = *(const v8bf*)(pn + fo);
            v8bf nhi = *(const v8bf*)(pn + fo + 8);
            accr  = wmma_bf16(alo, ahi, rlo, rhi, accr);
            accz  = wmma_bf16(alo, ahi, zlo, zhi, accz);
            accnx = wmma_bf16(alo, ahi, nlo, nhi, accnx);
        }
        // h-part of K (k0 = 16..47): gate-n accumulates into accnh.
        #pragma unroll 4
        for (int k0 = KSTEPS_X; k0 < KSTEPS; ++k0) {
            const int kk = k0 * 32;
            const size_t fo = (size_t)k0 * FRAG_ELEMS;
            v8bf alo = *(const v8bf*)(arow + kk + koff);
            v8bf ahi = *(const v8bf*)(arow + kk + 16 + koff);
            v8bf rlo = *(const v8bf*)(pr + fo);
            v8bf rhi = *(const v8bf*)(pr + fo + 8);
            v8bf zlo = *(const v8bf*)(pz + fo);
            v8bf zhi = *(const v8bf*)(pz + fo + 8);
            v8bf nlo = *(const v8bf*)(pn + fo);
            v8bf nhi = *(const v8bf*)(pn + fo + 8);
            accr  = wmma_bf16(alo, ahi, rlo, rhi, accr);
            accz  = wmma_bf16(alo, ahi, zlo, zhi, accz);
            accnh = wmma_bf16(alo, ahi, nlo, nhi, accnh);
        }

        // Epilogue: r/z/n for element (m, j) sit in the same lane & slot across
        // the four accumulators -> pure VALU, no cross-lane traffic.
        #pragma unroll
        for (int i = 0; i < 8; ++i) {
            const int ml = (lane >> 4) * 8 + i;     // row within tile
            const int m  = m0 + ml;                 // batch index
            float hp = (float)lds_a[ml * LDS_STRIDE + I_DIM + j];  // h_prev(m, j)
            float r  = fast_sigmoid(accr[i] + brc);
            float z  = fast_sigmoid(accz[i] + bzc);
            float n  = fast_tanh(accnx[i] + bnx + r * (accnh[i] + bnh));
            float hv = (1.0f - z) * n + z * hp;
            out[((size_t)t * B_DIM + m) * H_DIM + j] = hv;
            hnext[(size_t)m * H_DIM + j] = (bf16_t)hv;
            if (t == T_LEN - 1)
                out[(size_t)T_LEN * B_DIM * H_DIM + (size_t)m * H_DIM + j] = hv;
        }

        // Grid-wide barrier: release h_next, wait for all 32 blocks.
        __syncthreads();
        if (tid == 0) {
            __threadfence();
            atomicAdd(ctr, 1u);
            const unsigned target = (unsigned)NBLOCKS * (unsigned)(t + 1);
            while (__hip_atomic_load(ctr, __ATOMIC_RELAXED, __HIP_MEMORY_SCOPE_AGENT) < target)
                __builtin_amdgcn_s_sleep(1);
            __threadfence();
        }
        __syncthreads();
    }
}

// ---------------------------------------------------------------------------

extern "C" void kernel_launch(void* const* d_in, const int* in_sizes, int n_in,
                              void* d_out, int out_size, void* d_ws, size_t ws_size,
                              hipStream_t stream) {
    const float* x   = (const float*)d_in[0];   // (T, B, I)
    const float* wih = (const float*)d_in[1];   // (3H, I)
    const float* whh = (const float*)d_in[2];   // (3H, H)
    const float* bih = (const float*)d_in[3];   // (3H)
    const float* bhh = (const float*)d_in[4];   // (3H)
    float* out = (float*)d_out;                 // (T,B,H) ++ (1,B,H)

    char* ws = (char*)d_ws;
    unsigned* ctr  = (unsigned*)(ws + CTR_OFF);
    bf16_t*   hbuf = (bf16_t*)(ws + HBUF_OFF);
    bf16_t*   wpk  = (bf16_t*)(ws + WPK_OFF);
    bf16_t*   xbf  = (bf16_t*)(ws + XBF_OFF);

    // Re-initialize per call (graph-capture safe, deterministic across replays).
    hipMemsetAsync(ctr, 0, 256, stream);
    hipMemsetAsync(hbuf, 0, HBUF_ONE, stream);   // h0 = 0 (ping buffer)

    cvt_x_kernel<<<4096, NTHREADS, 0, stream>>>(x, xbf);
    build_wpk_kernel<<<2048, NTHREADS, 0, stream>>>(wih, whh, wpk);

    gru_scan_kernel<<<NBLOCKS, NTHREADS, 0, stream>>>(xbf, wpk, bih, bhh,
                                                      out, hbuf, ctr);
}
